// Seq2Seq_90520730730535
// MI455X (gfx1250) — compile-verified, matrix operand order checked by reading, not run on hardware
//
#include <hip/hip_runtime.h>
#include <hip/hip_bf16.h>

// ---------------------------------------------------------------------------
// Seq2Seq decoder for MI455X (gfx1250, wave32, WMMA).
// Encoder output is dead code in the reference -> skipped entirely.
// 128 WGs x 256 threads (8 waves); each WG owns 32 batch rows for all 30 steps.
// Each wave owns 32 hidden units and BOTH 16-row subtiles (B-fragment reuse x2).
// 8 waves/WG = 2 waves/SIMD -> large per-wave VGPR budget (no scratch spills;
// compiler uses the extended VGPR file via s_set_vgpr_msb).
// GEMMs via v_wmma_f32_16x16x32_bf16; outputs stored non-temporally.
// ---------------------------------------------------------------------------

typedef __bf16 bf16_t;
typedef __attribute__((ext_vector_type(16))) __bf16 v16bf;
typedef __attribute__((ext_vector_type(8)))  float  v8f;

#define B_TOTAL   4096
#define A_DIM     256
#define T_DIM     128
#define STEPS     30
#define START_TOK 1
#define M_TILE    32      // batch rows per workgroup
#define THREADS   256     // 8 waves
#define HP        264     // padded bf16 row stride for h  (256 + 8)
#define XP        392     // padded bf16 row stride for xin (384 + 8)
#define NT_GATE   64      // 1024 gate cols / 16
#define NT_LIN    16      // 256 cols / 16

union BFragU { uint4 u[2]; v16bf v; };

__device__ __forceinline__ v16bf load_bfrag_global(const bf16_t* base, int tile, int lane) {
    BFragU f;
    const uint4* p = (const uint4*)(base + (size_t)tile * 512) + lane * 2;
    f.u[0] = p[0];
    f.u[1] = p[1];
    return f.v;
}

// A fragment (16x32 bf16): lane l -> row = l%16, K base = kt*32 + (l/16)*8;
// VGPR0-3 hold K = kb..kb+7, VGPR4-7 hold K = kb+16..kb+23 (contiguous runs).
__device__ __forceinline__ v16bf load_afrag_lds(const bf16_t* rows, int stride, int kt, int lane) {
    int m  = lane & 15;
    int kb = kt * 32 + ((lane >> 4) << 3);
    const bf16_t* r = rows + m * stride;
    BFragU f;
    f.u[0] = *(const uint4*)(r + kb);
    f.u[1] = *(const uint4*)(r + kb + 16);
    return f.v;
}

__device__ __forceinline__ float sigmoidf_fast(float x) {
    return __builtin_amdgcn_rcpf(1.0f + __expf(-x));
}
__device__ __forceinline__ float tanhf_fast(float x) {
    float e = __expf(2.0f * x);
    return (e - 1.0f) * __builtin_amdgcn_rcpf(e + 1.0f);
}

#define WMMA_BF16(A, Bm, C) \
    __builtin_amdgcn_wmma_f32_16x16x32_bf16(false, (A), false, (Bm), (short)0, (C), false, false)

// One LSTM layer: this wave handles 32 rows x 32 hidden units (2 row subtiles).
// gates = xin @ W^T + h_rec @ U^T + bias ; torch gate order i,f,g,o.
// bv[u][g] are this lane's per-column bias values (hoisted, loop-invariant).
__device__ __forceinline__ void lstm_layer(
    const bf16_t* __restrict__ xin, int xin_stride, int xin_kt,
    const bf16_t* __restrict__ hrec,
    const bf16_t* __restrict__ Wpk, const bf16_t* __restrict__ Upk,
    const float (*bv)[4],
    float* cst,                // [32] per-lane cell state: [msub][u][r]
    bf16_t* __restrict__ hout,
    int ngroup, int lane)
{
    const v8f vzero = {0.f,0.f,0.f,0.f,0.f,0.f,0.f,0.f};
    v8f acc[2][4][2];
#pragma unroll
    for (int ms = 0; ms < 2; ++ms)
#pragma unroll
        for (int g = 0; g < 4; ++g)
#pragma unroll
            for (int u = 0; u < 2; ++u) acc[ms][g][u] = vzero;

#pragma unroll 1
    for (int kt = 0; kt < xin_kt; ++kt) {
        v16bf a0 = load_afrag_lds(xin, xin_stride, kt, lane);
        v16bf a1 = load_afrag_lds(xin + 16 * xin_stride, xin_stride, kt, lane);
#pragma unroll
        for (int g = 0; g < 4; ++g)
#pragma unroll
            for (int u = 0; u < 2; ++u) {
                v16bf b = load_bfrag_global(Wpk, kt * NT_GATE + g * 16 + ngroup * 2 + u, lane);
                acc[0][g][u] = WMMA_BF16(a0, b, acc[0][g][u]);
                acc[1][g][u] = WMMA_BF16(a1, b, acc[1][g][u]);
            }
    }
#pragma unroll 1
    for (int kt = 0; kt < 8; ++kt) {
        v16bf a0 = load_afrag_lds(hrec, HP, kt, lane);
        v16bf a1 = load_afrag_lds(hrec + 16 * HP, HP, kt, lane);
#pragma unroll
        for (int g = 0; g < 4; ++g)
#pragma unroll
            for (int u = 0; u < 2; ++u) {
                v16bf b = load_bfrag_global(Upk, kt * NT_GATE + g * 16 + ngroup * 2 + u, lane);
                acc[0][g][u] = WMMA_BF16(a0, b, acc[0][g][u]);
                acc[1][g][u] = WMMA_BF16(a1, b, acc[1][g][u]);
            }
    }
    __syncthreads();   // all waves done READING h before anyone overwrites it

    const int mrow  = (lane >> 4) << 3;          // D tile: M = 8*(lane/16) + r
    const int col16 = lane & 15;                 //         N = lane%16
#pragma unroll
    for (int u = 0; u < 2; ++u) {
        const int col = ngroup * 32 + u * 16 + col16;
        const float bi = bv[u][0];
        const float bf = bv[u][1];
        const float bg = bv[u][2];
        const float bo = bv[u][3];
#pragma unroll
        for (int ms = 0; ms < 2; ++ms)
#pragma unroll
            for (int r = 0; r < 8; ++r) {
                float i  = sigmoidf_fast(acc[ms][0][u][r] + bi);
                float f  = sigmoidf_fast(acc[ms][1][u][r] + bf);
                float gg = tanhf_fast   (acc[ms][2][u][r] + bg);
                float o  = sigmoidf_fast(acc[ms][3][u][r] + bo);
                float c  = f * cst[ms * 16 + u * 8 + r] + i * gg;
                cst[ms * 16 + u * 8 + r] = c;
                hout[(ms * 16 + mrow + r) * HP + col] = (bf16_t)(o * tanhf_fast(c));
            }
    }
}

__global__ __launch_bounds__(THREADS)
__attribute__((amdgpu_waves_per_eu(2)))
void decoder_kernel(
    const float* __restrict__ tags,
    const bf16_t* __restrict__ pkW0, const bf16_t* __restrict__ pkU0,
    const bf16_t* __restrict__ pkW1, const bf16_t* __restrict__ pkU1,
    const bf16_t* __restrict__ pkW2, const bf16_t* __restrict__ pkU2,
    const bf16_t* __restrict__ pkLin,
    const float* __restrict__ biases,     // [b0 1024 | b1 1024 | b2 1024 | blin 256]
    float* __restrict__ out)
{
    __shared__ alignas(16) bf16_t sh_h[3][M_TILE][HP];
    __shared__ alignas(16) bf16_t sh_x[M_TILE][XP];   // [prob(256) | tags(128)]

    const int tid    = threadIdx.x;
    const int lane   = tid & 31;
    const int ngroup = tid >> 5;                      // wave id = hidden-unit group
    const int wgBase = blockIdx.x * M_TILE;
    const int col16  = lane & 15;

    // ---- init: h=0, xin = [one-hot(START) | tags], write step-0 probs ----
    for (int i = tid; i < 3 * M_TILE * HP; i += THREADS)
        (&sh_h[0][0][0])[i] = (bf16_t)0.0f;
    for (int i = tid; i < M_TILE * A_DIM; i += THREADS) {
        int m = i >> 8, col = i & 255;
        float v = (col == START_TOK) ? 1.0f : 0.0f;
        sh_x[m][col] = (bf16_t)v;
        __builtin_nontemporal_store(v, &out[((size_t)(wgBase + m) * 31) * 256 + col]);
    }
    for (int i = tid; i < M_TILE * T_DIM; i += THREADS) {
        int m = i >> 7, j = i & 127;
        sh_x[m][256 + j] = (bf16_t)tags[(size_t)(wgBase + m) * T_DIM + j];
    }
    __syncthreads();

    float c0[32], c1[32], c2[32];
#pragma unroll
    for (int i = 0; i < 32; ++i) { c0[i] = 0.f; c1[i] = 0.f; c2[i] = 0.f; }

    // ---- hoist per-lane bias values (loop-invariant across steps) ----
    float bias_r[3][2][4];   // [layer][u][gate]
#pragma unroll
    for (int L = 0; L < 3; ++L)
#pragma unroll
        for (int u = 0; u < 2; ++u)
#pragma unroll
            for (int g = 0; g < 4; ++g)
                bias_r[L][u][g] = biases[L * 1024 + g * 256 + ngroup * 32 + u * 16 + col16];
    float biasL_r[2];
#pragma unroll
    for (int u = 0; u < 2; ++u)
        biasL_r[u] = biases[3072 + ngroup * 32 + u * 16 + col16];

    const size_t outIdxBase = (size_t)B_TOTAL * 31 * 256;
    const v8f vzero = {0.f,0.f,0.f,0.f,0.f,0.f,0.f,0.f};

    // parallel-argmax assignment: 8 lanes per batch row, 32 columns each
    const int amRow = ngroup * 4 + (lane >> 3);   // 0..31
    const int amCg  = lane & 7;                   // column group

#pragma unroll 1
    for (int step = 0; step < STEPS; ++step) {
        lstm_layer(&sh_x[0][0], XP, 12, &sh_h[0][0][0], pkW0, pkU0, bias_r[0], c0,
                   &sh_h[0][0][0], ngroup, lane);
        __syncthreads();
        lstm_layer(&sh_h[0][0][0], HP, 8, &sh_h[1][0][0], pkW1, pkU1, bias_r[1], c1,
                   &sh_h[1][0][0], ngroup, lane);
        __syncthreads();
        lstm_layer(&sh_h[1][0][0], HP, 8, &sh_h[2][0][0], pkW2, pkU2, bias_r[2], c2,
                   &sh_h[2][0][0], ngroup, lane);
        __syncthreads();

        // ---- linear (256x256) + sigmoid ----
        v8f acc[2][2];
        acc[0][0] = vzero; acc[0][1] = vzero; acc[1][0] = vzero; acc[1][1] = vzero;
        const bf16_t* hrows = &sh_h[2][0][0];
#pragma unroll 1
        for (int kt = 0; kt < 8; ++kt) {
            v16bf a0 = load_afrag_lds(hrows, HP, kt, lane);
            v16bf a1 = load_afrag_lds(hrows + 16 * HP, HP, kt, lane);
#pragma unroll
            for (int u = 0; u < 2; ++u) {
                v16bf b = load_bfrag_global(pkLin, kt * NT_LIN + ngroup * 2 + u, lane);
                acc[0][u] = WMMA_BF16(a0, b, acc[0][u]);
                acc[1][u] = WMMA_BF16(a1, b, acc[1][u]);
            }
        }
        const int mrow = (lane >> 4) << 3;
#pragma unroll
        for (int u = 0; u < 2; ++u) {
            const int col = ngroup * 32 + u * 16 + col16;
            const float bb = biasL_r[u];
#pragma unroll
            for (int ms = 0; ms < 2; ++ms)
#pragma unroll
                for (int r = 0; r < 8; ++r) {
                    float p = sigmoidf_fast(acc[ms][u][r] + bb);
                    int m = ms * 16 + mrow + r;
                    __builtin_nontemporal_store(
                        p, &out[((size_t)(wgBase + m) * 31 + (step + 1)) * 256 + col]);
                    sh_x[m][col] = (bf16_t)p;    // feeds next step's layer 0
                }
        }
        __syncthreads();

        // ---- per-row argmax, parallel: 8 lanes/row scan 32 cols, shfl-reduce ----
        {
            float bvv = -1.0f; int best = 0;
#pragma unroll 1
            for (int j = 0; j < 32; ++j) {
                int c = amCg * 32 + j;
                float v = (float)sh_x[amRow][c];
                if (v > bvv) { bvv = v; best = c; }
            }
#pragma unroll
            for (int mask = 1; mask < 8; mask <<= 1) {
                float ov = __shfl_xor(bvv, mask);
                int   oi = __shfl_xor(best, mask);
                if (ov > bvv || (ov == bvv && oi < best)) { bvv = ov; best = oi; }
            }
            if (amCg == 0)
                __builtin_nontemporal_store(
                    (float)best, &out[outIdxBase + (size_t)(wgBase + amRow) * STEPS + step]);
        }
        __syncthreads();
    }
}

// Pack W[n][k] (row-major, gates = x @ W^T) into WMMA bf16 B-fragment tiles:
// tile (kt,nt) holds B[k][n] for k in [kt*32,kt*32+32), n in [nt*16,nt*16+16);
// per lane 16 contiguous bf16 in exact fragment order.
__global__ void pack_kernel(const float* __restrict__ W, bf16_t* __restrict__ out,
                            int K, int Ntiles)
{
    int lane = threadIdx.x >> 4;          // 0..31
    int j    = threadIdx.x & 15;          // 0..15
    int kt   = blockIdx.x, nt = blockIdx.y;
    int n  = nt * 16 + (lane & 15);
    int kb = kt * 32 + ((lane >> 4) << 3);
    int k  = kb + j + ((j >= 8) ? 8 : 0); // runs kb..kb+7 and kb+16..kb+23
    size_t tile = (size_t)kt * Ntiles + nt;
    out[tile * 512 + lane * 16 + j] = (bf16_t)W[(size_t)n * K + k];
}

__global__ void bias_kernel(const float* __restrict__ bih0, const float* __restrict__ bhh0,
                            const float* __restrict__ bih,  const float* __restrict__ bhh,
                            const float* __restrict__ linb, float* __restrict__ out)
{
    int t = threadIdx.x;                  // 1024 threads
    out[t]        = bih0[t] + bhh0[t];
    out[1024 + t] = bih[t] + bhh[t];
    out[2048 + t] = bih[1024 + t] + bhh[1024 + t];
    if (t < 256) out[3072 + t] = linb[t];
}

extern "C" void kernel_launch(void* const* d_in, const int* in_sizes, int n_in,
                              void* d_out, int out_size, void* d_ws, size_t ws_size,
                              hipStream_t stream)
{
    (void)in_sizes; (void)n_in; (void)out_size; (void)ws_size;
    // setup_inputs order (encoder + embedding inputs are dead code):
    const float* tags    = (const float*)d_in[2];
    const float* l2_Wih0 = (const float*)d_in[13];   // [1024][384]
    const float* l2_Whh0 = (const float*)d_in[14];   // [1024][256]
    const float* l2_bih0 = (const float*)d_in[15];
    const float* l2_bhh0 = (const float*)d_in[16];
    const float* l2_Wih  = (const float*)d_in[17];   // [2][1024][256]
    const float* l2_Whh  = (const float*)d_in[18];   // [2][1024][256]
    const float* l2_bih  = (const float*)d_in[19];   // [2][1024]
    const float* l2_bhh  = (const float*)d_in[20];
    const float* lin_W   = (const float*)d_in[21];   // [256][256]
    const float* lin_b   = (const float*)d_in[22];

    bf16_t* ws = (bf16_t*)d_ws;
    size_t off = 0;
    bf16_t* pkW0 = ws + off; off += (size_t)12 * 64 * 512;
    bf16_t* pkU0 = ws + off; off += (size_t)8  * 64 * 512;
    bf16_t* pkW1 = ws + off; off += (size_t)8  * 64 * 512;
    bf16_t* pkU1 = ws + off; off += (size_t)8  * 64 * 512;
    bf16_t* pkW2 = ws + off; off += (size_t)8  * 64 * 512;
    bf16_t* pkU2 = ws + off; off += (size_t)8  * 64 * 512;
    bf16_t* pkLn = ws + off; off += (size_t)8  * 16 * 512;
    float*  biases = (float*)(ws + off);             // +13 KB, total ~3.55 MB

    pack_kernel<<<dim3(12, 64), 512, 0, stream>>>(l2_Wih0, pkW0, 384, 64);
    pack_kernel<<<dim3(8,  64), 512, 0, stream>>>(l2_Whh0, pkU0, 256, 64);
    pack_kernel<<<dim3(8,  64), 512, 0, stream>>>(l2_Wih,              pkW1, 256, 64);
    pack_kernel<<<dim3(8,  64), 512, 0, stream>>>(l2_Whh,              pkU1, 256, 64);
    pack_kernel<<<dim3(8,  64), 512, 0, stream>>>(l2_Wih + 1024 * 256, pkW2, 256, 64);
    pack_kernel<<<dim3(8,  64), 512, 0, stream>>>(l2_Whh + 1024 * 256, pkU2, 256, 64);
    pack_kernel<<<dim3(8,  16), 512, 0, stream>>>(lin_W, pkLn, 256, 16);
    bias_kernel<<<1, 1024, 0, stream>>>(l2_bih0, l2_bhh0, l2_bih, l2_bhh, lin_b, biases);

    decoder_kernel<<<B_TOTAL / M_TILE, THREADS, 0, stream>>>(
        tags, pkW0, pkU0, pkW1, pkU1, pkW2, pkU2, pkLn, biases, (float*)d_out);
}
